// PlurallisticAlignmentModule_74053826118321
// MI455X (gfx1250) — matrix operand order repaired and checked by reading, not compiled, not run on hardware
//
#include <hip/hip_runtime.h>
#include <hip/hip_bf16.h>

// ---------------------------------------------------------------------------
// Sizes fixed by the reference: B=32768, C=1024, H=1024, S=8, DC=64, Kext=2048
// ---------------------------------------------------------------------------

typedef __bf16  v16bf __attribute__((ext_vector_type(16)));
typedef float   v8f   __attribute__((ext_vector_type(8)));
typedef int     i32x4 __attribute__((ext_vector_type(4)));

#ifndef __has_builtin
#define __has_builtin(x) 0
#endif

#if __has_builtin(__builtin_amdgcn_global_load_async_to_lds_b128) && \
    __has_builtin(__builtin_amdgcn_s_wait_asynccnt)
#define USE_ASYNC_LDS 1
#else
#define USE_ASYNC_LDS 0
#endif

__device__ __forceinline__ unsigned short f2bf(float f) {
    unsigned u = __builtin_bit_cast(unsigned, f);
    unsigned r = 0x7FFFu + ((u >> 16) & 1u);          // round-to-nearest-even
    return (unsigned short)((u + r) >> 16);
}
__device__ __forceinline__ float bf2f(unsigned short h) {
    unsigned u = ((unsigned)h) << 16;
    return __builtin_bit_cast(float, u);
}

union FragB16 { v16bf v; unsigned long long q[4]; unsigned short h[16]; };
union AccF    { v8f   v; float f[8]; };

// 16-bf16 (32B) tile staging: async DMA to LDS when available, else sync copy.
// Builtin signature (from compiler diagnostic): param0 = int4 in global AS.
__device__ __forceinline__ void stage16(unsigned short* lds_dst,
                                        const unsigned short* gsrc)
{
#if USE_ASYNC_LDS
    typedef __attribute__((address_space(1))) i32x4 gi4;
    typedef __attribute__((address_space(3))) i32x4 li4;
    __builtin_amdgcn_global_load_async_to_lds_b128(
        (gi4*)gsrc, (li4*)lds_dst, 0, 0);
    __builtin_amdgcn_global_load_async_to_lds_b128(
        (gi4*)(gsrc + 8), (li4*)(lds_dst + 8), 0, 0);
#else
    const unsigned long long* s = (const unsigned long long*)gsrc;
    unsigned long long d0 = s[0], d1 = s[1], d2 = s[2], d3 = s[3];
    unsigned long long* d = (unsigned long long*)lds_dst;
    d[0] = d0; d[1] = d1; d[2] = d2; d[3] = d3;
#endif
}

__device__ __forceinline__ void stage_wait()
{
#if USE_ASYNC_LDS
    __builtin_amdgcn_s_wait_asynccnt(0);
#endif
}

// ---------------------------------------------------------------------------
// Kernel 1: extended activation  Aext[b] = [logits[b] | style_emb[idx[b]]] bf16
// ---------------------------------------------------------------------------
__global__ void k_build_aext(const float* __restrict__ logits,
                             const int*   __restrict__ sidx,
                             const float* __restrict__ emb,
                             unsigned short* __restrict__ out,
                             long long total)
{
    long long i = (long long)blockIdx.x * blockDim.x + threadIdx.x;
    if (i >= total) return;
    long long b   = i >> 11;          // Kext = 2048
    int       col = (int)(i & 2047);
    float v;
    if (col < 1024) v = logits[(b << 10) + col];
    else            v = emb[((size_t)sidx[b] << 10) + (col - 1024)];
    out[i] = f2bf(v);
}

// ---------------------------------------------------------------------------
// Kernel 2: one-time weight transpose + f32->bf16:  out[n][k] = bf16(in[k][n])
// LDS-tiled so both global accesses are coalesced. Block 32x8, tile 32x32.
// ---------------------------------------------------------------------------
__global__ __launch_bounds__(256)
void k_cvt_transpose_bf16(const float* __restrict__ in,        // K x N
                          unsigned short* __restrict__ out,    // N x K
                          int K, int N)
{
    __shared__ float tile[32][33];
    const int nb = blockIdx.x * 32;
    const int kb = blockIdx.y * 32;
    const int tx = threadIdx.x & 31;
    const int ty = threadIdx.x >> 5;     // 0..7
#pragma unroll
    for (int i = 0; i < 32; i += 8)
        tile[ty + i][tx] = in[(size_t)(kb + ty + i) * N + nb + tx];
    __syncthreads();
#pragma unroll
    for (int i = 0; i < 32; i += 8)
        out[(size_t)(nb + ty + i) * K + kb + tx] = f2bf(tile[tx][ty + i]);
}

// ---------------------------------------------------------------------------
// Kernel 3: bf16 WMMA GEMM.  Block tile 128x128, K-step 32, 256 threads =
// 8 wave32 waves arranged 2(M) x 4(N); wave tile 64x32 = 4x2 16x16 WMMAs.
// A: MxK bf16 row-major.  Wt: NxK bf16 (pre-transposed) so BOTH tiles stage
// with identical vectorized 32B copies and B fragments read contiguous K.
// Double-buffered LDS; staging via global_load_async_to_lds_b128 when present.
//   mode 0: out_bf16 = relu(A@W + bias)
//   mode 1: out_f32  = A@W + bias + residual
// ---------------------------------------------------------------------------
#define BM 128
#define BN 128
#define BK 32
#define LDSS 40            // 80B rows: 16B-aligned chunks, conflict-free reads

__global__ __launch_bounds__(256)
void k_gemm_bf16(const unsigned short* __restrict__ A,
                 const unsigned short* __restrict__ Wt,
                 const float* __restrict__ bias,
                 const float* __restrict__ residual,
                 unsigned short* __restrict__ outb,
                 float* __restrict__ outf,
                 int M, int N, int K, int mode)
{
    __shared__ unsigned short As[2][BM * LDSS];
    __shared__ unsigned short Bs[2][BN * LDSS];   // [n][k]

    const int tid   = threadIdx.x;
    const int lane  = tid & 31;
    const int wid   = tid >> 5;
    const int wm    = wid >> 2;      // 0..1
    const int wn    = wid & 3;       // 0..3
    const int lrow  = lane & 15;
    const int lhalf = lane >> 4;     // 0/1

    const int bN = blockIdx.x * BN;
    const int bM = blockIdx.y * BM;

    AccF acc[4][2];
#pragma unroll
    for (int i = 0; i < 4; ++i)
#pragma unroll
        for (int j = 0; j < 2; ++j)
#pragma unroll
            for (int r = 0; r < 8; ++r) acc[i][j].f[r] = 0.f;

    // staging: thread t owns row t>>1 (of A tile and of B tile), 16-elem half
    const int sRow  = tid >> 1;      // 0..127
    const int sHalf = tid & 1;

    const unsigned short* aSrc = A  + (size_t)(bM + sRow) * K + sHalf * 16;
    const unsigned short* bSrc = Wt + (size_t)(bN + sRow) * K + sHalf * 16;
    const int ldsOff = sRow * LDSS + sHalf * 16;

    // prologue: stage tile 0 into buffer 0
    stage16(&As[0][ldsOff], aSrc);
    stage16(&Bs[0][ldsOff], bSrc);
    stage_wait();
    __syncthreads();

    const int ktiles = K / BK;
    for (int t = 0; t < ktiles; ++t) {
        const int cur = t & 1;
        // kick off next tile into the other buffer while we compute
        if (t + 1 < ktiles) {
            stage16(&As[cur ^ 1][ldsOff], aSrc + (t + 1) * BK);
            stage16(&Bs[cur ^ 1][ldsOff], bSrc + (t + 1) * BK);
        }

        // A 16x32 frag: lanes 0-15 row M=lane hold K 0..7 & 16..23,
        //               lanes16-31 row M=lane-16 hold K 8..15 & 24..31.
        FragB16 af[4];
#pragma unroll
        for (int i = 0; i < 4; ++i) {
            const int m = wm * 64 + i * 16 + lrow;
            const unsigned long long* p =
                (const unsigned long long*)(&As[cur][m * LDSS + lhalf * 8]);
            af[i].q[0] = p[0]; af[i].q[1] = p[1];
            const unsigned long long* p2 =
                (const unsigned long long*)(&As[cur][m * LDSS + 16 + lhalf * 8]);
            af[i].q[2] = p2[0]; af[i].q[3] = p2[1];
        }
        // B 32x16 frag: lanes 0-15 col N=lane hold K 0..7 & 16..23,
        //               lanes16-31 hold K 8..15 & 24..31.
        FragB16 bfr[2];
#pragma unroll
        for (int j = 0; j < 2; ++j) {
            const int n = wn * 32 + j * 16 + lrow;
            const unsigned long long* p =
                (const unsigned long long*)(&Bs[cur][n * LDSS + lhalf * 8]);
            bfr[j].q[0] = p[0]; bfr[j].q[1] = p[1];
            const unsigned long long* p2 =
                (const unsigned long long*)(&Bs[cur][n * LDSS + 16 + lhalf * 8]);
            bfr[j].q[2] = p2[0]; bfr[j].q[3] = p2[1];
        }

#pragma unroll
        for (int i = 0; i < 4; ++i)
#pragma unroll
            for (int j = 0; j < 2; ++j)
                acc[i][j].v = __builtin_amdgcn_wmma_f32_16x16x32_bf16(
                    false, af[i].v, false, bfr[j].v,
                    (short)0, acc[i][j].v, false, false);

        stage_wait();        // our async copies for tile t+1 are complete
        __syncthreads();     // everyone done reading buf[cur] & copies visible
    }

    // epilogue: C/D layout -> VGPR r holds rows {r, r+8}, col = lane&15
#pragma unroll
    for (int i = 0; i < 4; ++i) {
#pragma unroll
        for (int j = 0; j < 2; ++j) {
            const int col = bN + wn * 32 + j * 16 + lrow;
            const float bv = bias[col];
#pragma unroll
            for (int r = 0; r < 8; ++r) {
                const int row = bM + wm * 64 + i * 16 + lhalf * 8 + r;
                const size_t off = (size_t)row * N + col;
                float v = acc[i][j].f[r] + bv;
                if (mode == 0) {
                    v = fmaxf(v, 0.f);
                    outb[off] = f2bf(v);
                } else {
                    v += residual[off];
                    outf[off] = v;
                }
            }
        }
    }
}

// ---------------------------------------------------------------------------
// Kernel 4: row softmax (N=1024), 256 threads/row, probs out in bf16
// ---------------------------------------------------------------------------
__global__ __launch_bounds__(256)
void k_softmax_rows(const float* __restrict__ x,
                    unsigned short* __restrict__ p, int N)
{
    const int b   = blockIdx.x;
    const int tid = threadIdx.x;
    const float* row = x + (size_t)b * N;

    float vals[4];
    float m = -3.402823466e38f;
#pragma unroll
    for (int i = 0; i < 4; ++i) {
        vals[i] = row[tid + i * 256];
        m = fmaxf(m, vals[i]);
    }
#pragma unroll
    for (int off = 16; off > 0; off >>= 1)
        m = fmaxf(m, __shfl_xor(m, off));

    __shared__ float redm[8];
    __shared__ float reds[8];
    if ((tid & 31) == 0) redm[tid >> 5] = m;
    __syncthreads();
    float m0 = redm[0];
#pragma unroll
    for (int w = 1; w < 8; ++w) m0 = fmaxf(m0, redm[w]);

    float s = 0.f;
#pragma unroll
    for (int i = 0; i < 4; ++i) {
        vals[i] = __expf(vals[i] - m0);
        s += vals[i];
    }
#pragma unroll
    for (int off = 16; off > 0; off >>= 1)
        s += __shfl_xor(s, off);
    if ((tid & 31) == 0) reds[tid >> 5] = s;
    __syncthreads();
    float stot = 0.f;
#pragma unroll
    for (int w = 0; w < 8; ++w) stot += reds[w];
    const float inv = 1.0f / stot;

    unsigned short* prow = p + (size_t)b * N;
#pragma unroll
    for (int i = 0; i < 4; ++i)
        prow[tid + i * 256] = f2bf(vals[i] * inv);
}

// ---------------------------------------------------------------------------
// Kernel 5: per-sample selected-style calibrator
//   conf[b] = sigmoid( relu(probs_b @ Wc1[s] + bc1[s]) . Wc2[s] + bc2[s] )
// ---------------------------------------------------------------------------
__global__ __launch_bounds__(64)
void k_calib(const unsigned short* __restrict__ probs,
             const int* __restrict__ sidx,
             const float* __restrict__ Wc1, const float* __restrict__ bc1,
             const float* __restrict__ Wc2, const float* __restrict__ bc2,
             float* __restrict__ conf, int C, int DC)
{
    const int b = blockIdx.x;
    const int d = threadIdx.x;            // 0..63
    const int s = sidx[b];

    const float* w = Wc1 + ((size_t)s * C) * DC;
    const unsigned short* pr = probs + (size_t)b * C;   // block-uniform

    float h = 0.f;
    for (int c = 0; c < C; ++c)
        h += bf2f(pr[c]) * w[(size_t)c * DC + d];
    h += bc1[s * DC + d];
    h = fmaxf(h, 0.f);

    float t = h * Wc2[s * DC + d];
#pragma unroll
    for (int off = 16; off > 0; off >>= 1)
        t += __shfl_down(t, off);

    __shared__ float partial[2];
    if ((d & 31) == 0) partial[d >> 5] = t;
    __syncthreads();
    if (d == 0) {
        float sum = partial[0] + partial[1] + bc2[s];
        conf[b] = 1.0f / (1.0f + __expf(-sum));
    }
}

// ---------------------------------------------------------------------------
// launch
// ---------------------------------------------------------------------------
extern "C" void kernel_launch(void* const* d_in, const int* in_sizes, int n_in,
                              void* d_out, int out_size, void* d_ws, size_t ws_size,
                              hipStream_t stream)
{
    const float* logits    = (const float*)d_in[0];
    const int*   sidx      = (const int*)  d_in[1];
    const float* style_emb = (const float*)d_in[2];
    const float* Wa1       = (const float*)d_in[3];
    const float* ba1       = (const float*)d_in[4];
    const float* Wa2       = (const float*)d_in[5];
    const float* ba2       = (const float*)d_in[6];
    const float* Wc1       = (const float*)d_in[7];
    const float* bc1       = (const float*)d_in[8];
    const float* Wc2       = (const float*)d_in[9];
    const float* bc2       = (const float*)d_in[10];

    const int Bn = 32768, C = 1024, H = 1024, DC = 64;
    const int Kext = C + H;   // 2048

    // workspace layout (bf16 buffers)
    unsigned short* aext   = (unsigned short*)d_ws;            // Bn * Kext
    unsigned short* wt1    = aext   + (size_t)Bn * Kext;       // H x Kext (transposed Wa1)
    unsigned short* wt2    = wt1    + (size_t)H * Kext;        // C x H    (transposed Wa2)
    unsigned short* hidden = wt2    + (size_t)C * H;           // Bn * H
    unsigned short* probs  = hidden + (size_t)Bn * H;          // Bn * C

    float* adjusted = (float*)d_out;                 // Bn * C
    float* conf     = adjusted + (size_t)Bn * C;     // Bn

    // 1) extended activation (gather + concat + f32->bf16)
    {
        long long total = (long long)Bn * Kext;
        int blocks = (int)((total + 255) / 256);
        k_build_aext<<<blocks, 256, 0, stream>>>(logits, sidx, style_emb, aext, total);
    }
    // 2) one-time weight transpose + downconvert
    {
        dim3 g1(H / 32, Kext / 32);          // Wa1: (2048 x 1024) -> (1024 x 2048)
        k_cvt_transpose_bf16<<<g1, 256, 0, stream>>>(Wa1, wt1, Kext, H);
        dim3 g2(C / 32, H / 32);             // Wa2: (1024 x 1024) -> (1024 x 1024)
        k_cvt_transpose_bf16<<<g2, 256, 0, stream>>>(Wa2, wt2, H, C);
    }
    // 3) GEMM1: hidden = relu(aext @ Wa1 + ba1)   [32768 x 2048 x 1024]
    {
        dim3 grid(H / BN, Bn / BM);   // (8, 256)
        k_gemm_bf16<<<grid, 256, 0, stream>>>(aext, wt1, ba1, nullptr,
                                              hidden, nullptr,
                                              Bn, H, Kext, 0);
    }
    // 4) GEMM2: adjusted = logits + hidden @ Wa2 + ba2   [32768 x 1024 x 1024]
    {
        dim3 grid(C / BN, Bn / BM);   // (8, 256)
        k_gemm_bf16<<<grid, 256, 0, stream>>>(hidden, wt2, ba2, logits,
                                              nullptr, adjusted,
                                              Bn, C, H, 1);
    }
    // 5) softmax rows -> probs (bf16)
    k_softmax_rows<<<Bn, 256, 0, stream>>>(adjusted, probs, C);

    // 6) selected-style calibrator -> conf
    k_calib<<<Bn, 64, 0, stream>>>(probs, sidx, Wc1, bc1, Wc2, bc2, conf, C, DC);
}
